// GNNRegressor_56229711839628
// MI455X (gfx1250) — matrix operand order repaired and checked by reading, not compiled, not run on hardware
//
#include <hip/hip_runtime.h>
#include <hip/hip_bf16.h>
#include <hip/hip_fp16.h>

typedef __attribute__((ext_vector_type(16))) _Float16 v16h;
typedef __attribute__((ext_vector_type(8)))  _Float16 v8h;
typedef __attribute__((ext_vector_type(4)))  _Float16 v4h;
typedef __attribute__((ext_vector_type(8)))  float    v8f;

#define HD 128
#define LAYERS 4
#define NGRAPH 512

// ---------------- utility ----------------
__global__ void zero_f32(float* __restrict__ p, int n) {
    int i = blockIdx.x * blockDim.x + threadIdx.x;
    if (i < n) p[i] = 0.f;
}

// ---------------- prep: node table T[120][128] = relu(node_emb @ W_lin + b_lin) ----------------
__global__ void node_table_kernel(const float* __restrict__ node_emb,
                                  const float* __restrict__ W_lin,
                                  const float* __restrict__ b_lin,
                                  float* __restrict__ T) {
    int r = blockIdx.x;       // 0..119
    int j = threadIdx.x;      // 0..127
    float s = b_lin[j];
    for (int k = 0; k < HD; ++k)
        s = fmaf(node_emb[r * HD + k], W_lin[k * HD + j], s);
    T[r * HD + j] = fmaxf(s, 0.f);
}

// ---------------- prep: sigma[60] = sigmoid(edge_emb @ W_el + b_el) ----------------
__global__ void sigma_table_kernel(const float* __restrict__ edge_emb,
                                   const float* __restrict__ W_el,
                                   const float* __restrict__ b_el,
                                   float* __restrict__ sigma) {
    int a = threadIdx.x;
    if (a >= 60) return;
    float s = b_el[0];
    for (int k = 0; k < HD; ++k)
        s = fmaf(edge_emb[a * HD + k], W_el[k], s);
    sigma[a] = 1.f / (1.f + expf(-s));
}

// ---------------- prep: shuffle gcn_W into WMMA B-fragment order ----------------
// Bfrag layout: [layer][ntile(8)][kstep(4)][lane(32)][j(16 f16)]
// lane l: N = t*16 + (l&15), half = l>>4 ; element j: K = ks*32 + (j&7) + 8*half + 16*(j>>3)
__global__ void wfrag_kernel(const float* __restrict__ gcn_W, _Float16* __restrict__ Bfrag) {
    int idx = blockIdx.x * blockDim.x + threadIdx.x;
    if (idx >= LAYERS * 8 * 4 * 32 * 16) return;
    int j    = idx & 15;
    int lane = (idx >> 4) & 31;
    int ks   = (idx >> 9) & 3;
    int t    = (idx >> 11) & 7;
    int i    = idx >> 14;
    int half = lane >> 4;
    int N = t * 16 + (lane & 15);
    int K = ks * 32 + (j & 7) + 8 * half + 16 * (j >> 3);
    Bfrag[idx] = (_Float16)gcn_W[(i * HD + K) * HD + N];
}

// ---------------- node init: h = T[x[n]], also f16 copy ----------------
__global__ void node_init_kernel(const int* __restrict__ x,
                                 const float* __restrict__ T,
                                 float* __restrict__ h,
                                 _Float16* __restrict__ h16, int n_nodes) {
    int idx = blockIdx.x * blockDim.x + threadIdx.x;    // over N*32 float4 chunks
    if (idx >= n_nodes * 32) return;
    int n = idx >> 5, q = idx & 31;
    float4 v = ((const float4*)(T + (size_t)x[n] * HD))[q];
    ((float4*)(h + (size_t)n * HD))[q] = v;
    v4h hv; hv[0] = (_Float16)v.x; hv[1] = (_Float16)v.y; hv[2] = (_Float16)v.z; hv[3] = (_Float16)v.w;
    *(v4h*)(h16 + (size_t)n * HD + q * 4) = hv;
}

// ---------------- edge weights + degree ----------------
__global__ void edge_w_kernel(const int* __restrict__ edge_attr,
                              const int* __restrict__ col,
                              const float* __restrict__ sigma,
                              float* __restrict__ w,
                              float* __restrict__ deg, int n_edges) {
    int e = blockIdx.x * blockDim.x + threadIdx.x;
    if (e >= n_edges) return;
    float wv = sigma[edge_attr[e]];
    w[e] = wv;
    unsafeAtomicAdd(&deg[col[e]], wv);
}

__global__ void dis_kernel(const float* __restrict__ deg, float* __restrict__ dis, int n_nodes) {
    int n = blockIdx.x * blockDim.x + threadIdx.x;
    if (n >= n_nodes) return;
    float d = deg[n];
    dis[n] = (d > 0.f) ? rsqrtf(fmaxf(d, 1e-12f)) : 0.f;
}

__global__ void norm_kernel(const int* __restrict__ row, const int* __restrict__ col,
                            const float* __restrict__ w, const float* __restrict__ dis,
                            float* __restrict__ nrm, int n_edges) {
    int e = blockIdx.x * blockDim.x + threadIdx.x;
    if (e >= n_edges) return;
    nrm[e] = dis[row[e]] * w[e] * dis[col[e]];
}

// ---------------- GEMM: m = h @ W[i]  (WMMA f16->f32), also agg := bias ----------------
__global__ void gcn_gemm_kernel(const _Float16* __restrict__ h16,
                                const _Float16* __restrict__ Bfrag,   // this layer: [8][4][32][16]
                                const float* __restrict__ bias,       // gcn_b[i]
                                float* __restrict__ m,
                                float* __restrict__ agg, int n_rows) {
    int wave = threadIdx.x >> 5;
    int lane = threadIdx.x & 31;
    int row0 = (blockIdx.x * (blockDim.x >> 5) + wave) * 16;
    if (row0 >= n_rows) return;
    int mr = row0 + (lane & 15);
    int half = lane >> 4;

    // A fragments for all 4 K-steps (ISA 16-bit A 16x32 layout)
    v16h a[4];
    const _Float16* hrow = h16 + (size_t)mr * HD;
#pragma unroll
    for (int ks = 0; ks < 4; ++ks) {
        const _Float16* p = hrow + ks * 32 + half * 8;
        v8h lo = *(const v8h*)(p);
        v8h hi = *(const v8h*)(p + 16);
        a[ks] = __builtin_shufflevector(lo, hi, 0,1,2,3,4,5,6,7,8,9,10,11,12,13,14,15);
    }

#pragma unroll
    for (int t = 0; t < 8; ++t) {
        v8f c = {};
#pragma unroll
        for (int ks = 0; ks < 4; ++ks) {
            v16h b = *(const v16h*)(Bfrag + (((t * 4 + ks) * 32 + lane) << 4));
            c = __builtin_amdgcn_wmma_f32_16x16x32_f16(false, a[ks], false, b,
                                                       (short)0, c, false, false);
        }
        int n = t * 16 + (lane & 15);
        float bv = bias[n];
#pragma unroll
        for (int r = 0; r < 8; ++r) {
            int rr = row0 + r + 8 * half;           // C/D layout: VGPR r, halves split M by 8
            m[(size_t)rr * HD + n] = c[r];
            agg[(size_t)rr * HD + n] = bv;          // init accumulator with bias
        }
    }
}

// ---------------- scatter: agg[col] += norm * m[row]  (one edge per wave) ----------------
__global__ void scatter_kernel(const int* __restrict__ row, const int* __restrict__ col,
                               const float* __restrict__ nrm,
                               const float* __restrict__ m,
                               float* __restrict__ agg, int n_edges) {
    int wave = threadIdx.x >> 5;
    int lane = threadIdx.x & 31;
    int e = blockIdx.x * (blockDim.x >> 5) + wave;
    if (e >= n_edges) return;
    float nv = nrm[e];
    int rs = row[e], cd = col[e];
    float4 mv = ((const float4*)(m + (size_t)rs * HD))[lane];
    float* ap = agg + (size_t)cd * HD + lane * 4;
    unsafeAtomicAdd(ap + 0, nv * mv.x);
    unsafeAtomicAdd(ap + 1, nv * mv.y);
    unsafeAtomicAdd(ap + 2, nv * mv.z);
    unsafeAtomicAdd(ap + 3, nv * mv.w);
}

// ---------------- update: h = LN(relu(agg) + h) ; one wave per node ----------------
__global__ void update_kernel(const float* __restrict__ agg,
                              const float* __restrict__ ln_g,
                              const float* __restrict__ ln_b,
                              float* __restrict__ h,
                              _Float16* __restrict__ h16, int n_nodes) {
    int wave = threadIdx.x >> 5;
    int lane = threadIdx.x & 31;
    int n = blockIdx.x * (blockDim.x >> 5) + wave;
    if (n >= n_nodes) return;
    float4 a4 = ((const float4*)(agg + (size_t)n * HD))[lane];
    float4 h4 = ((const float4*)(h   + (size_t)n * HD))[lane];
    float t0 = fmaxf(a4.x, 0.f) + h4.x;
    float t1 = fmaxf(a4.y, 0.f) + h4.y;
    float t2 = fmaxf(a4.z, 0.f) + h4.z;
    float t3 = fmaxf(a4.w, 0.f) + h4.w;
    float s  = t0 + t1 + t2 + t3;
    float sq = t0*t0 + t1*t1 + t2*t2 + t3*t3;
#pragma unroll
    for (int off = 1; off < 32; off <<= 1) {        // wave32 butterfly
        s  += __shfl_xor(s,  off, 32);
        sq += __shfl_xor(sq, off, 32);
    }
    float mu  = s * (1.f / HD);
    float var = sq * (1.f / HD) - mu * mu;
    float inv = rsqrtf(var + 1e-5f);
    int k = lane * 4;
    float o0 = (t0 - mu) * inv * ln_g[k+0] + ln_b[k+0];
    float o1 = (t1 - mu) * inv * ln_g[k+1] + ln_b[k+1];
    float o2 = (t2 - mu) * inv * ln_g[k+2] + ln_b[k+2];
    float o3 = (t3 - mu) * inv * ln_g[k+3] + ln_b[k+3];
    ((float4*)(h + (size_t)n * HD))[lane] = make_float4(o0, o1, o2, o3);
    v4h hv; hv[0] = (_Float16)o0; hv[1] = (_Float16)o1; hv[2] = (_Float16)o2; hv[3] = (_Float16)o3;
    *(v4h*)(h16 + (size_t)n * HD + k) = hv;
}

// ---------------- pooling: pool[batch[n]] += h[n] ----------------
__global__ void pool_add_kernel(const int* __restrict__ batch,
                                const float* __restrict__ h,
                                float* __restrict__ pool, int n_nodes) {
    int idx = blockIdx.x * blockDim.x + threadIdx.x;    // N*32 float4 chunks
    if (idx >= n_nodes * 32) return;
    int n = idx >> 5, q = idx & 31;
    float4 v = ((const float4*)(h + (size_t)n * HD))[q];
    float* pp = pool + (size_t)batch[n] * HD + q * 4;
    unsafeAtomicAdd(pp + 0, v.x);
    unsafeAtomicAdd(pp + 1, v.y);
    unsafeAtomicAdd(pp + 2, v.z);
    unsafeAtomicAdd(pp + 3, v.w);
}

// ---------------- readout MLP: one block (64 threads) per graph ----------------
__global__ void mlp_kernel(const float* __restrict__ pool,
                           const float* __restrict__ Wfc1, const float* __restrict__ bfc1,
                           const float* __restrict__ g1,   const float* __restrict__ bb1,
                           const float* __restrict__ Wfc2, const float* __restrict__ bfc2,
                           const float* __restrict__ g2,   const float* __restrict__ bb2,
                           const float* __restrict__ Wfc3, const float* __restrict__ bfc3,
                           float* __restrict__ out) {
    __shared__ float gin[HD];
    __shared__ float s1[64];
    __shared__ float s2[32];
    const float BN = 0.9999950000374996f;   // 1/sqrt(1+1e-5)
    int gid = blockIdx.x;
    int t = threadIdx.x;                     // 64 threads
    gin[t]      = pool[(size_t)gid * HD + t];
    gin[t + 64] = pool[(size_t)gid * HD + t + 64];
    __syncthreads();
    float a = bfc1[t];
    for (int k = 0; k < HD; ++k) a = fmaf(gin[k], Wfc1[k * 64 + t], a);
    s1[t] = fmaxf(a, 0.f) * (g1[t] * BN) + bb1[t];
    __syncthreads();
    if (t < 32) {
        float a2 = bfc2[t];
        for (int k = 0; k < 64; ++k) a2 = fmaf(s1[k], Wfc2[k * 32 + t], a2);
        s2[t] = fmaxf(a2, 0.f) * (g2[t] * BN) + bb2[t];
    }
    __syncthreads();
    if (t == 0) {
        float a3 = bfc3[0];
        for (int k = 0; k < 32; ++k) a3 = fmaf(s2[k], Wfc3[k], a3);
        out[gid] = a3;
    }
}

// =====================================================================
extern "C" void kernel_launch(void* const* d_in, const int* in_sizes, int n_in,
                              void* d_out, int out_size, void* d_ws, size_t ws_size,
                              hipStream_t stream) {
    const int N = in_sizes[0];
    const int E = in_sizes[2];
    const int G = NGRAPH;

    const int*   x          = (const int*)d_in[0];
    const int*   edge_index = (const int*)d_in[1];
    const int*   edge_attr  = (const int*)d_in[2];
    const int*   batch      = (const int*)d_in[3];
    const float* node_emb   = (const float*)d_in[5];
    const float* edge_emb   = (const float*)d_in[6];
    const float* W_el       = (const float*)d_in[7];
    const float* b_el       = (const float*)d_in[8];
    const float* W_lin      = (const float*)d_in[9];
    const float* b_lin      = (const float*)d_in[10];
    const float* gcn_W      = (const float*)d_in[11];
    const float* gcn_b      = (const float*)d_in[12];
    const float* ln_g       = (const float*)d_in[13];
    const float* ln_b       = (const float*)d_in[14];
    const float* W_fc1      = (const float*)d_in[15];
    const float* b_fc1      = (const float*)d_in[16];
    const float* bn1_g      = (const float*)d_in[17];
    const float* bn1_b      = (const float*)d_in[18];
    const float* W_fc2      = (const float*)d_in[19];
    const float* b_fc2      = (const float*)d_in[20];
    const float* bn2_g      = (const float*)d_in[21];
    const float* bn2_b      = (const float*)d_in[22];
    const float* W_fc3      = (const float*)d_in[23];
    const float* b_fc3      = (const float*)d_in[24];
    float* out = (float*)d_out;

    const int* row = edge_index;        // [0, :]
    const int* col = edge_index + E;    // [1, :]

    // ---- workspace carve-up ----
    char* ws = (char*)d_ws;
    size_t off = 0;
    auto carve = [&](size_t bytes) -> char* {
        char* p = ws + off;
        off = (off + bytes + 255) & ~(size_t)255;
        return p;
    };
    float*    h     = (float*)   carve((size_t)N * HD * 4);
    _Float16* h16   = (_Float16*)carve((size_t)N * HD * 2);
    float*    m     = (float*)   carve((size_t)N * HD * 4);
    float*    agg   = (float*)   carve((size_t)N * HD * 4);
    float*    w     = (float*)   carve((size_t)E * 4);
    float*    nrm   = (float*)   carve((size_t)E * 4);
    float*    deg   = (float*)   carve((size_t)N * 4);
    float*    dis   = (float*)   carve((size_t)N * 4);
    float*    T     = (float*)   carve(120 * HD * 4);
    float*    sigma = (float*)   carve(64 * 4);
    _Float16* Bfrag = (_Float16*)carve((size_t)LAYERS * 8 * 4 * 32 * 16 * 2);
    float*    pool  = (float*)   carve((size_t)G * HD * 4);
    (void)ws_size;

    // ---- prep (tiny) ----
    node_table_kernel<<<120, 128, 0, stream>>>(node_emb, W_lin, b_lin, T);
    sigma_table_kernel<<<1, 64, 0, stream>>>(edge_emb, W_el, b_el, sigma);
    wfrag_kernel<<<(LAYERS * 8 * 4 * 32 * 16 + 255) / 256, 256, 0, stream>>>(gcn_W, Bfrag);
    zero_f32<<<(N + 255) / 256, 256, 0, stream>>>(deg, N);
    zero_f32<<<(G * HD + 255) / 256, 256, 0, stream>>>(pool, G * HD);

    // ---- graph preprocessing ----
    node_init_kernel<<<((size_t)N * 32 + 255) / 256, 256, 0, stream>>>(x, T, h, h16, N);
    edge_w_kernel<<<(E + 255) / 256, 256, 0, stream>>>(edge_attr, col, sigma, w, deg, E);
    dis_kernel<<<(N + 255) / 256, 256, 0, stream>>>(deg, dis, N);
    norm_kernel<<<(E + 255) / 256, 256, 0, stream>>>(row, col, w, dis, nrm, E);

    // ---- GCN layers ----
    int strips = (N + 15) / 16;
    int gemm_blocks = (strips + 7) / 8;            // 8 waves/block, 1 strip/wave
    int scat_blocks = (E + 7) / 8;                 // 1 edge/wave
    int upd_blocks  = (N + 7) / 8;                 // 1 node/wave
    for (int i = 0; i < LAYERS; ++i) {
        const _Float16* Bi = Bfrag + (size_t)i * 8 * 4 * 32 * 16;
        gcn_gemm_kernel<<<gemm_blocks, 256, 0, stream>>>(h16, Bi, gcn_b + i * HD, m, agg, N);
        scatter_kernel<<<scat_blocks, 256, 0, stream>>>(row, col, nrm, m, agg, E);
        update_kernel<<<upd_blocks, 256, 0, stream>>>(agg, ln_g + i * HD, ln_b + i * HD, h, h16, N);
    }

    // ---- readout ----
    pool_add_kernel<<<((size_t)N * 32 + 255) / 256, 256, 0, stream>>>(batch, h, pool, N);
    mlp_kernel<<<G, 64, 0, stream>>>(pool, W_fc1, b_fc1, bn1_g, bn1_b,
                                     W_fc2, b_fc2, bn2_g, bn2_b, W_fc3, b_fc3, out);
}
